// SSD_65635690217961
// MI455X (gfx1250) — matrix operand order repaired and checked by reading, not compiled, not run on hardware
//
#include <hip/hip_runtime.h>
#include <math.h>

typedef __attribute__((ext_vector_type(16))) __bf16 v16bf;
typedef __attribute__((ext_vector_type(8)))  __bf16 v8bf;
typedef __attribute__((ext_vector_type(8)))  float  v8f;

constexpr int  BATCH       = 32;
constexpr long TOTAL_BOXES = 8732;
constexpr int  NCLS        = 81;                              // K+1
constexpr long CONF_ELEMS  = (long)BATCH * TOTAL_BOXES * NCLS;

static __device__ __forceinline__ v16bf zero16() {
  v16bf z;
  #pragma unroll
  for (int e = 0; e < 16; ++e) z[e] = (__bf16)0.f;
  return z;
}

// ---------------------------------------------------------------------------
// One-time layout transforms (all buffers fit in L2; cost ~= one pass of HBM)
// ---------------------------------------------------------------------------

// feat NCHW f32 -> NHWC bf16
__global__ void nchw2nhwc_bf16(const float* __restrict__ X, __bf16* __restrict__ Y,
                               int C, int HW, int total) {
  int p = blockIdx.x * blockDim.x + threadIdx.x;
  if (p >= total) return;
  int nb = p / HW, pix = p - nb * HW;
  const float* xp = X + (size_t)nb * C * HW + pix;
  __bf16* yp = Y + (size_t)p * C;
  for (int c = 0; c < C; c += 8) {
    v8bf o;
    #pragma unroll
    for (int u = 0; u < 8; ++u) o[u] = (__bf16)xp[(size_t)(c + u) * HW];
    *(v8bf*)(yp + c) = o;
  }
}

// L2Norm over channels fused with NCHW f32 -> NHWC bf16 transform
__global__ void l2norm_nhwc_bf16(const float* __restrict__ X, const float* __restrict__ Sc,
                                 __bf16* __restrict__ Y, int C, int HW, int total) {
  int p = blockIdx.x * blockDim.x + threadIdx.x;
  if (p >= total) return;
  int nb = p / HW, pix = p - nb * HW;
  const float* xp = X + (size_t)nb * C * HW + pix;
  float ss = 0.f;
  for (int c = 0; c < C; ++c) { float v = xp[(size_t)c * HW]; ss = fmaf(v, v, ss); }
  float inv = 1.0f / (sqrtf(ss) + 1e-10f);
  __bf16* yp = Y + (size_t)p * C;
  for (int c = 0; c < C; c += 8) {
    v8bf o;
    #pragma unroll
    for (int u = 0; u < 8; ++u)
      o[u] = (__bf16)(Sc[c + u] * xp[(size_t)(c + u) * HW] * inv);
    *(v8bf*)(yp + c) = o;
  }
}

// OIHW f32 weights -> [Co][KHKW*Ci] bf16 with k' = (r*KW+t)*Ci + ci
__global__ void wtrans_bf16(const float* __restrict__ Wsrc, __bf16* __restrict__ Wdst,
                            int Ci, int KHKW, int K, int total) {
  int idx = blockIdx.x * blockDim.x + threadIdx.x;
  if (idx >= total) return;
  int co = idx / K;
  int kp = idx - co * K;
  int rt = kp / Ci;
  int ci = kp - rt * Ci;
  Wdst[idx] = (__bf16)Wsrc[((size_t)co * Ci + ci) * KHKW + rt];
}

// ---------------------------------------------------------------------------
// Implicit-GEMM conv: X NHWC bf16, Wb K-permuted bf16, f32 accumulate.
// Wave computes a 32(M) x 32(N) supertile = 2x2 WMMA tiles; the tap loop (r,t)
// is fully hoisted (Ci % 32 == 0 for every conv here), so the hot loop is:
//   8 x b128 loads + 4 x v_wmma_f32_16x16x32_bf16 per 32-wide K step.
// Fragment layouts per CDNA5 ISA 7.12.2:
//   A: lane m=lane&15, half=lane>>4; e<8 -> K=8*half+e ; e>=8 -> K=16+8*half+e-8
//   B: lane n=lane&15; same K mapping as A   (contiguous ci in NHWC!)
//   C/D: vgpr r -> M = r + 8*half, N = lane&15  (8 consecutive channels/lane)
// ---------------------------------------------------------------------------
template<int KH, int KW, int S, int P, bool RELU, bool HEAD>
__global__ __launch_bounds__(256) void conv_wmma_kernel(
    const __bf16* __restrict__ X, const __bf16* __restrict__ Wb,
    const float* __restrict__ Bias, void* __restrict__ Yv,
    int Ci, int Co, int H, int W, int Ho, int Wo,
    int Npix, int stM, int totalST,
    int mbox, int cls, long baseBox)
{
  int wid = blockIdx.x * 8 + (threadIdx.x >> 5);
  if (wid >= totalST) return;                   // wave-uniform exit
  const int lane = threadIdx.x & 31;
  const int col  = lane & 15;
  const int half = lane >> 4;
  const int tm = wid % stM;
  const int tn = wid / stM;
  const int m0 = tm * 32;
  const int q0 = tn * 32;
  const int K  = Ci * KH * KW;

  // ---- N subtiles: decompose pixel columns once ----
  int nbv[2], hov[2], wov[2]; bool qokv[2];
  #pragma unroll
  for (int j = 0; j < 2; ++j) {
    int q = q0 + 16 * j + col;
    qokv[j] = q < Npix;
    int nb = 0, ho = 0, wo = 0;
    if (qokv[j]) {
      int how = Ho * Wo;
      nb = q / how; int rem = q - nb * how;
      ho = rem / Wo; wo = rem - ho * Wo;
    }
    nbv[j] = nb; hov[j] = ho; wov[j] = wo;
  }
  // ---- M subtiles: weight row pointers ----
  const __bf16* arow[2]; bool mokv[2];
  #pragma unroll
  for (int i = 0; i < 2; ++i) {
    int m = m0 + 16 * i + col;
    mokv[i] = m < Co;
    arow[i] = Wb + (size_t)(mokv[i] ? m : 0) * K;
  }

  v8f acc[2][2];
  #pragma unroll
  for (int i = 0; i < 2; ++i)
    #pragma unroll
    for (int j = 0; j < 2; ++j)
      #pragma unroll
      for (int r = 0; r < 8; ++r) acc[i][j][r] = 0.f;

  const v16bf zf = zero16();

  #pragma unroll
  for (int rt = 0; rt < KH * KW; ++rt) {        // tap loop, fully hoisted
    const int r = rt / KW, t = rt - r * KW;     // compile-time constants
    // per-tap input row pointers per N subtile
    const __bf16* bptr[2]; bool bok[2];
    #pragma unroll
    for (int j = 0; j < 2; ++j) {
      int hh = hov[j] * S - P + r;
      int ww = wov[j] * S - P + t;
      bok[j] = qokv[j] && (unsigned)hh < (unsigned)H && (unsigned)ww < (unsigned)W;
      bptr[j] = X + (((size_t)nbv[j] * H + (bok[j] ? hh : 0)) * W + (bok[j] ? ww : 0)) * Ci;
    }
    const __bf16* ap[2] = { arow[0] + rt * Ci, arow[1] + rt * Ci };

    for (int c0 = 0; c0 < Ci; c0 += 32) {       // hot loop
      v16bf A[2], Bf[2];
      #pragma unroll
      for (int i = 0; i < 2; ++i) {
        if (mokv[i]) {
          v8bf lo = *(const v8bf*)(ap[i] + c0 + 8 * half);
          v8bf hi = *(const v8bf*)(ap[i] + c0 + 16 + 8 * half);
          A[i] = __builtin_shufflevector(lo, hi, 0,1,2,3,4,5,6,7,8,9,10,11,12,13,14,15);
        } else A[i] = zf;
      }
      #pragma unroll
      for (int j = 0; j < 2; ++j) {
        if (bok[j]) {
          const __bf16* bp = bptr[j] + c0 + 8 * half;
          v8bf lo = *(const v8bf*)(bp);
          v8bf hi = *(const v8bf*)(bp + 16);
          Bf[j] = __builtin_shufflevector(lo, hi, 0,1,2,3,4,5,6,7,8,9,10,11,12,13,14,15);
        } else Bf[j] = zf;
      }
      #pragma unroll
      for (int i = 0; i < 2; ++i)
        #pragma unroll
        for (int j = 0; j < 2; ++j)
          acc[i][j] = __builtin_amdgcn_wmma_f32_16x16x32_bf16(
              false, A[i], false, Bf[j], (short)0, acc[i][j], false, false);
    }
  }

  // ---- epilogue ----
  #pragma unroll
  for (int j = 0; j < 2; ++j) {
    if (!qokv[j]) continue;
    const long po = ((long)nbv[j] * Ho + hov[j]) * Wo + wov[j];
    #pragma unroll
    for (int i = 0; i < 2; ++i) {
      const int mmBase = m0 + 16 * i + 8 * half;   // 8 consecutive channels
      if (HEAD) {
        float* Y = (float*)Yv;
        #pragma unroll
        for (int r = 0; r < 8; ++r) {
          const int mm = mmBase + r;
          if (mm < Co) {
            float v = acc[i][j][r] + Bias[mm];
            const int  aIdx = mm / cls;
            const int  cIdx = mm - aIdx * cls;
            const long box  = baseBox + (long)(hov[j] * Wo + wov[j]) * mbox + aIdx;
            Y[((long)nbv[j] * TOTAL_BOXES + box) * cls + cIdx] = v;
          }
        }
      } else {
        // intermediates: Co is a multiple of 16 -> single aligned b128 store
        if (mmBase < Co) {
          __bf16* Y = (__bf16*)Yv;
          v8bf o;
          #pragma unroll
          for (int r = 0; r < 8; ++r) {
            float v = acc[i][j][r] + Bias[mmBase + r];
            if (RELU) v = fmaxf(v, 0.f);
            o[r] = (__bf16)v;
          }
          *(v8bf*)(Y + po * Co + mmBase) = o;
        }
      }
    }
  }
}

// ---------------------------------------------------------------------------
template<int KH, int KW, int S, int P, bool RELU, bool HEAD>
static void run_conv(const __bf16* X, const __bf16* Wb, const float* Bias, void* Y,
                     int Ci, int Co, int H, int W, int Ho, int Wo,
                     int mbox, int cls, long baseBox, hipStream_t stream) {
  const int Npix    = BATCH * Ho * Wo;
  const int stM     = (Co + 31) / 32;
  const int stN     = (Npix + 31) / 32;
  const int totalST = stM * stN;
  const int blocks  = (totalST + 7) / 8;         // 8 waves / 256-thread block
  conv_wmma_kernel<KH, KW, S, P, RELU, HEAD>
      <<<dim3(blocks), dim3(256), 0, stream>>>(
          X, Wb, Bias, Y, Ci, Co, H, W, Ho, Wo,
          Npix, stM, totalST, mbox, cls, baseBox);
}

static void run_wtrans(const float* Wsrc, __bf16* Wdst, int Co, int Ci, int KHKW,
                       hipStream_t stream) {
  const int K = Ci * KHKW;
  const int total = Co * K;
  wtrans_bf16<<<dim3((total + 255) / 256), dim3(256), 0, stream>>>(
      Wsrc, Wdst, Ci, KHKW, K, total);
}

extern "C" void kernel_launch(void* const* d_in, const int* in_sizes, int n_in,
                              void* d_out, int out_size, void* d_ws, size_t ws_size,
                              hipStream_t stream) {
  (void)in_sizes; (void)n_in; (void)out_size; (void)ws_size;
  const float* feat0    = (const float*)d_in[0];
  const float* feat1    = (const float*)d_in[1];
  const float* l2norm_w = (const float*)d_in[2];
  const float* extra_w[8]; const float* extra_b[8];
  for (int i = 0; i < 8; ++i) { extra_w[i] = (const float*)d_in[3 + i];
                                extra_b[i] = (const float*)d_in[11 + i]; }
  const float* cls_w[6]; const float* cls_b[6];
  const float* box_w[6]; const float* box_b[6];
  for (int i = 0; i < 6; ++i) { cls_w[i] = (const float*)d_in[19 + i];
                                cls_b[i] = (const float*)d_in[25 + i];
                                box_w[i] = (const float*)d_in[31 + i];
                                box_b[i] = (const float*)d_in[37 + i]; }
  float* out = (float*)d_out;

  // ---- workspace carve (all bf16; every size is a multiple of 8 elems) ----
  __bf16* p = (__bf16*)d_ws;
  auto carve = [&](long n) { __bf16* r = p; p += n; return r; };
  // activations, NHWC bf16
  __bf16* f0 = carve((long)BATCH * 38 * 38 * 512);
  __bf16* x1 = carve((long)BATCH * 19 * 19 * 1024);
  __bf16* e0 = carve((long)BATCH * 19 * 19 * 256);
  __bf16* e1 = carve((long)BATCH * 10 * 10 * 512);
  __bf16* e2 = carve((long)BATCH * 10 * 10 * 128);
  __bf16* e3 = carve((long)BATCH *  5 *  5 * 256);
  __bf16* e4 = carve((long)BATCH *  5 *  5 * 128);
  __bf16* e5 = carve((long)BATCH *  3 *  3 * 256);
  __bf16* e6 = carve((long)BATCH *  3 *  3 * 128);
  __bf16* e7 = carve((long)BATCH *  1 *  1 * 256);
  // weights, K-permuted bf16
  const int eCo[8] = {256,512,128,256,128,256,128,256};
  const int eCi[8] = {1024,256,512,128,256,128,256,128};
  const int eKK[8] = {1,9,1,9,1,9,1,9};
  __bf16* ew[8];
  for (int i = 0; i < 8; ++i) ew[i] = carve((long)eCo[i] * eCi[i] * eKK[i]);
  const int fCi[6]   = { 512, 1024, 512, 256, 256, 256 };
  const int fHW[6]   = { 38, 19, 10, 5, 3, 1 };
  const int fm[6]    = { 4, 6, 6, 6, 4, 4 };
  const long fbase[6]= { 0, 5776, 7942, 8542, 8692, 8728 };
  __bf16* cw[6]; __bf16* bw[6];
  for (int f = 0; f < 6; ++f) {
    cw[f] = carve((long)fm[f] * NCLS * fCi[f] * 9);
    bw[f] = carve((long)fm[f] * 4    * fCi[f] * 9);
  }

  // ---- 1) layout transforms ----
  {
    int total = BATCH * 38 * 38;
    l2norm_nhwc_bf16<<<dim3((total + 255) / 256), dim3(256), 0, stream>>>(
        feat0, l2norm_w, f0, 512, 38 * 38, total);
  }
  {
    int total = BATCH * 19 * 19;
    nchw2nhwc_bf16<<<dim3((total + 255) / 256), dim3(256), 0, stream>>>(
        feat1, x1, 1024, 19 * 19, total);
  }
  for (int i = 0; i < 8; ++i) run_wtrans(extra_w[i], ew[i], eCo[i], eCi[i], eKK[i], stream);
  for (int f = 0; f < 6; ++f) {
    run_wtrans(cls_w[f], cw[f], fm[f] * NCLS, fCi[f], 9, stream);
    run_wtrans(box_w[f], bw[f], fm[f] * 4,    fCi[f], 9, stream);
  }

  // ---- 2) extras chain (NHWC bf16 -> NHWC bf16, ReLU) ----
  run_conv<1,1,1,0,true,false>(x1, ew[0], extra_b[0], e0, 1024, 256, 19, 19, 19, 19, 0, 1, 0, stream);
  run_conv<3,3,2,1,true,false>(e0, ew[1], extra_b[1], e1,  256, 512, 19, 19, 10, 10, 0, 1, 0, stream);
  run_conv<1,1,1,0,true,false>(e1, ew[2], extra_b[2], e2,  512, 128, 10, 10, 10, 10, 0, 1, 0, stream);
  run_conv<3,3,2,1,true,false>(e2, ew[3], extra_b[3], e3,  128, 256, 10, 10,  5,  5, 0, 1, 0, stream);
  run_conv<1,1,1,0,true,false>(e3, ew[4], extra_b[4], e4,  256, 128,  5,  5,  5,  5, 0, 1, 0, stream);
  run_conv<3,3,1,0,true,false>(e4, ew[5], extra_b[5], e5,  128, 256,  5,  5,  3,  3, 0, 1, 0, stream);
  run_conv<1,1,1,0,true,false>(e5, ew[6], extra_b[6], e6,  256, 128,  3,  3,  3,  3, 0, 1, 0, stream);
  run_conv<3,3,1,0,true,false>(e6, ew[7], extra_b[7], e7,  128, 256,  3,  3,  1,  1, 0, 1, 0, stream);

  // ---- 3) heads: write straight into (conf || loc) ----
  const __bf16* fptr[6] = { f0, x1, e1, e3, e5, e7 };
  float* conf = out;
  float* loc  = out + CONF_ELEMS;
  for (int f = 0; f < 6; ++f) {
    const int d = fHW[f];
    run_conv<3,3,1,1,false,true>(fptr[f], cw[f], cls_b[f], conf,
                                 fCi[f], fm[f] * NCLS, d, d, d, d,
                                 fm[f], NCLS, fbase[f], stream);
    run_conv<3,3,1,1,false,true>(fptr[f], bw[f], box_b[f], loc,
                                 fCi[f], fm[f] * 4, d, d, d, d,
                                 fm[f], 4, fbase[f], stream);
  }
}
